// SinkhornOT_64372969832612
// MI455X (gfx1250) — compile-verified, matrix operand order checked by reading, not run on hardware
//
#include <hip/hip_runtime.h>
#include <hip/hip_bf16.h>

// Sinkhorn OT: N=16384 points, K=1024 prototypes, D=512.
// Phase 1: row norms + f32->f16 conversion (WMMA feed)
// Phase 2: C = dx + dp - 2*x@p^T via v_wmma_f32_16x16x32_f16 with register
//          double-buffering (loads overlap WMMA); C staged in d_out; global max
//          via deterministic uint atomicMax
// Phase 3: normalize C, Kmat = exp(-C/eps) (Kmat is 64MB -> L2-resident)
// Phase 4: 20 Sinkhorn iterations (row GEMV with LDS-staged v, split-column GEMV
//          with LDS-staged u slice, fixed-order reduces)
// Phase 5: T* = u*Kmat*v overwrites C in d_out; loss = sum(T*C) two-pass reduce.

#define NROWS 16384
#define KCOLS 1024
#define DDIM  512
#define EPS_INV 10.0f            // 1/0.1
#define MU (1.0f / 16384.0f)
#define NU (1.0f / 1024.0f)
#define RS 64                    // row slices for column GEMV (256 rows each)
#define LOSS_BLOCKS 4096

typedef __attribute__((ext_vector_type(16))) _Float16 v16h;
typedef __attribute__((ext_vector_type(8)))  _Float16 v8h;
typedef __attribute__((ext_vector_type(8)))  float    v8f;

union HFrag { v16h v; v8h h[2]; };

#define LOADF(f, base, off) do {                         \
    (f).h[0] = *(const v8h*)((base) + (off));            \
    (f).h[1] = *(const v8h*)((base) + (off) + 16);       \
} while (0)

#define WMMA(a, b, c) __builtin_amdgcn_wmma_f32_16x16x32_f16(false, (a).v, false, (b).v, (short)0, (c), false, false)

// ---- Phase 1: per-row squared norm + f16 conversion (one wave per row) ----
__global__ void sk_prep_rows(const float* __restrict__ src, _Float16* __restrict__ dst,
                             float* __restrict__ norms, int rows) {
    int wave = threadIdx.x >> 5;
    int lane = threadIdx.x & 31;
    int row  = blockIdx.x * 8 + wave;
    if (row >= rows) return;
    const float* s = src + (size_t)row * DDIM;
    _Float16*    d = dst + (size_t)row * DDIM;
    float acc = 0.0f;
    for (int c = lane; c < DDIM; c += 32) {
        float xv = s[c];
        acc += xv * xv;
        d[c] = (_Float16)xv;
    }
#pragma unroll
    for (int off = 16; off > 0; off >>= 1) acc += __shfl_xor(acc, off, 32);
    if (lane == 0) norms[row] = acc;
}

__global__ void sk_init(float* __restrict__ u, float* __restrict__ v,
                        unsigned* __restrict__ cmax_bits) {
    int i = blockIdx.x * blockDim.x + threadIdx.x;
    if (i < NROWS) u[i] = 1.0f;
    if (i < KCOLS) v[i] = 1.0f;
    if (i == 0) *cmax_bits = 0u;
}

// ---- Phase 2: WMMA GEMM -> cost matrix C, plus global max ----
__device__ __forceinline__ float sk_store_tile(float* __restrict__ Cout, const v8f acc,
                                               const float* dxv, float dpv,
                                               int m0, int half, int col) {
    float cm = 0.0f;
#pragma unroll
    for (int r = 0; r < 8; ++r) {
        int row = m0 + r + 8 * half;
        float c = dxv[r] + dpv - 2.0f * acc[r];
        Cout[(size_t)row * KCOLS + col] = c;
        cm = fmaxf(cm, c);
    }
    return cm;
}

__global__ void __launch_bounds__(256)
sk_gemm_cost(const _Float16* __restrict__ xh, const _Float16* __restrict__ ph,
             const float* __restrict__ dx, const float* __restrict__ dp,
             float* __restrict__ Cout, unsigned* __restrict__ cmax_bits) {
    __shared__ float red[256];
    int lane  = threadIdx.x & 31;
    int wave  = threadIdx.x >> 5;
    int laneM = lane & 15;
    int half  = lane >> 4;
    int m0    = blockIdx.y * 128 + wave * 16;   // 16 rows per wave
    int nblk  = blockIdx.x * 64;                // 64 cols per wave (4 tiles)

    v8f acc0 = {}, acc1 = {}, acc2 = {}, acc3 = {};
    // 16-bit A fragment: lanes 0-15 hold K=0..7 / 16..23, lanes 16-31 K=8..15 / 24..31
    const _Float16* abase = xh + (size_t)(m0 + laneM) * DDIM + half * 8;
    const _Float16* b0 = ph + (size_t)(nblk +  0 + laneM) * DDIM + half * 8;
    const _Float16* b1 = ph + (size_t)(nblk + 16 + laneM) * DDIM + half * 8;
    const _Float16* b2 = ph + (size_t)(nblk + 32 + laneM) * DDIM + half * 8;
    const _Float16* b3 = ph + (size_t)(nblk + 48 + laneM) * DDIM + half * 8;

    // Register double-buffering: ping (P) holds step d0, pong (Q) holds d0+32.
    HFrag aP, bP0, bP1, bP2, bP3;
    HFrag aQ, bQ0, bQ1, bQ2, bQ3;
    LOADF(aP, abase, 0);
    LOADF(bP0, b0, 0); LOADF(bP1, b1, 0); LOADF(bP2, b2, 0); LOADF(bP3, b3, 0);

#pragma unroll
    for (int d0 = 0; d0 < DDIM; d0 += 64) {
        // Prefetch pong while ping computes
        LOADF(aQ, abase, d0 + 32);
        LOADF(bQ0, b0, d0 + 32); LOADF(bQ1, b1, d0 + 32);
        LOADF(bQ2, b2, d0 + 32); LOADF(bQ3, b3, d0 + 32);

        acc0 = WMMA(aP, bP0, acc0);
        acc1 = WMMA(aP, bP1, acc1);
        acc2 = WMMA(aP, bP2, acc2);
        acc3 = WMMA(aP, bP3, acc3);

        if (d0 + 64 < DDIM) {   // compile-time resolved under full unroll
            LOADF(aP, abase, d0 + 64);
            LOADF(bP0, b0, d0 + 64); LOADF(bP1, b1, d0 + 64);
            LOADF(bP2, b2, d0 + 64); LOADF(bP3, b3, d0 + 64);
        }

        acc0 = WMMA(aQ, bQ0, acc0);
        acc1 = WMMA(aQ, bQ1, acc1);
        acc2 = WMMA(aQ, bQ2, acc2);
        acc3 = WMMA(aQ, bQ3, acc3);
    }

    float dxv[8];
#pragma unroll
    for (int r = 0; r < 8; ++r) dxv[r] = dx[m0 + r + 8 * half];

    float cm = 0.0f;
    cm = fmaxf(cm, sk_store_tile(Cout, acc0, dxv, dp[nblk +  0 + laneM], m0, half, nblk +  0 + laneM));
    cm = fmaxf(cm, sk_store_tile(Cout, acc1, dxv, dp[nblk + 16 + laneM], m0, half, nblk + 16 + laneM));
    cm = fmaxf(cm, sk_store_tile(Cout, acc2, dxv, dp[nblk + 32 + laneM], m0, half, nblk + 32 + laneM));
    cm = fmaxf(cm, sk_store_tile(Cout, acc3, dxv, dp[nblk + 48 + laneM], m0, half, nblk + 48 + laneM));

    red[threadIdx.x] = cm;
    __syncthreads();
#pragma unroll
    for (int s = 128; s > 0; s >>= 1) {
        if (threadIdx.x < s) red[threadIdx.x] = fmaxf(red[threadIdx.x], red[threadIdx.x + s]);
        __syncthreads();
    }
    // cm >= 0 so uint bit-pattern max == float max; max is order-independent -> deterministic
    if (threadIdx.x == 0) atomicMax(cmax_bits, __float_as_uint(red[0]));
}

// ---- Phase 3: normalize C (in d_out) and build Kmat = exp(-C/eps), float4 ----
__global__ void sk_expnorm(float4* __restrict__ C4, float4* __restrict__ K4,
                           const unsigned* __restrict__ cmax_bits) {
    float inv = 1.0f / (__uint_as_float(*cmax_bits) + 1e-8f);
    size_t idx    = (size_t)blockIdx.x * blockDim.x + threadIdx.x;
    size_t stride = (size_t)gridDim.x * blockDim.x;
    size_t total  = (size_t)NROWS * KCOLS / 4;
    for (; idx < total; idx += stride) {
        float4 c = C4[idx];
        c.x *= inv; c.y *= inv; c.z *= inv; c.w *= inv;
        C4[idx] = c;
        float4 kk;
        kk.x = __expf(-c.x * EPS_INV);
        kk.y = __expf(-c.y * EPS_INV);
        kk.z = __expf(-c.z * EPS_INV);
        kk.w = __expf(-c.w * EPS_INV);
        K4[idx] = kk;
    }
}

// ---- Phase 4: Sinkhorn iterations ----
__global__ void __launch_bounds__(256)
sk_row_update(const float* __restrict__ Kmat, const float* __restrict__ v,
              float* __restrict__ u) {
    __shared__ float vs[KCOLS];                       // 4KB: stage v once per block
    for (int k = threadIdx.x; k < KCOLS; k += 256) vs[k] = v[k];
    __syncthreads();

    int wave = threadIdx.x >> 5;
    int lane = threadIdx.x & 31;
    int row  = blockIdx.x * 8 + wave;
    const float4* kr4 = (const float4*)(Kmat + (size_t)row * KCOLS);
    float s = 0.0f;
    for (int q = lane; q < KCOLS / 4; q += 32) {       // float4 over 1024 cols
        float4 kv = kr4[q];
        const float4 vv = *(const float4*)&vs[q * 4];
        s += kv.x * vv.x + kv.y * vv.y + kv.z * vv.z + kv.w * vv.w;
    }
#pragma unroll
    for (int off = 16; off > 0; off >>= 1) s += __shfl_xor(s, off, 32);
    if (lane == 0) u[row] = MU / (s + 1e-8f);
}

__global__ void __launch_bounds__(256)
sk_col_partial(const float* __restrict__ Kmat, const float* __restrict__ u,
               float* __restrict__ part) {
    __shared__ float us[NROWS / RS];                  // 256 rows per slice
    int k  = blockIdx.x * 256 + threadIdx.x;
    int r0 = blockIdx.y * (NROWS / RS);
    us[threadIdx.x] = u[r0 + threadIdx.x];
    __syncthreads();

    float s = 0.0f;
    for (int i = 0; i < NROWS / RS; i += 4) {
        // prefetch 8 rows ahead (gfx1250 global_prefetch path)
        __builtin_prefetch(&Kmat[(size_t)(r0 + i + 8) * KCOLS + k], 0, 1);
        s += Kmat[(size_t)(r0 + i + 0) * KCOLS + k] * us[i + 0];
        s += Kmat[(size_t)(r0 + i + 1) * KCOLS + k] * us[i + 1];
        s += Kmat[(size_t)(r0 + i + 2) * KCOLS + k] * us[i + 2];
        s += Kmat[(size_t)(r0 + i + 3) * KCOLS + k] * us[i + 3];
    }
    part[(size_t)blockIdx.y * KCOLS + k] = s;
}

__global__ void sk_col_finish(const float* __restrict__ part, float* __restrict__ v) {
    int k = blockIdx.x * 256 + threadIdx.x;
    float s = 0.0f;
#pragma unroll 8
    for (int r = 0; r < RS; ++r) s += part[(size_t)r * KCOLS + k];
    v[k] = NU / (s + 1e-8f);
}

// ---- Phase 5: T* = u * Kmat * v (overwrites C in d_out) + loss partials ----
__global__ void __launch_bounds__(256)
sk_tstar(float4* __restrict__ CT4, const float4* __restrict__ K4,
         const float* __restrict__ u, const float* __restrict__ v,
         float* __restrict__ lpart) {
    __shared__ float red[256];
    size_t idx    = (size_t)blockIdx.x * blockDim.x + threadIdx.x;
    size_t stride = (size_t)gridDim.x * blockDim.x;
    size_t total  = (size_t)NROWS * KCOLS / 4;
    float s = 0.0f;
    for (; idx < total; idx += stride) {
        int row  = (int)(idx >> 8);                   // (idx*4) / 1024
        int col4 = (int)(idx & 255) * 4;
        float4 c  = CT4[idx];
        float4 kk = K4[idx];
        float  uv = u[row];
        const float4 vv = *(const float4*)&v[col4];
        float4 t;
        t.x = uv * kk.x * vv.x;
        t.y = uv * kk.y * vv.y;
        t.z = uv * kk.z * vv.z;
        t.w = uv * kk.w * vv.w;
        CT4[idx] = t;
        s += t.x * c.x + t.y * c.y + t.z * c.z + t.w * c.w;
    }
    red[threadIdx.x] = s;
    __syncthreads();
#pragma unroll
    for (int w = 128; w > 0; w >>= 1) {
        if (threadIdx.x < w) red[threadIdx.x] += red[threadIdx.x + w];
        __syncthreads();
    }
    if (threadIdx.x == 0) lpart[blockIdx.x] = red[0];
}

__global__ void sk_loss_finish(const float* __restrict__ lpart, float* __restrict__ out) {
    __shared__ float red[256];
    float s = 0.0f;
    for (int i = threadIdx.x; i < LOSS_BLOCKS; i += 256) s += lpart[i];
    red[threadIdx.x] = s;
    __syncthreads();
#pragma unroll
    for (int w = 128; w > 0; w >>= 1) {
        if (threadIdx.x < w) red[threadIdx.x] += red[threadIdx.x + w];
        __syncthreads();
    }
    if (threadIdx.x == 0) *out = red[0];
}

extern "C" void kernel_launch(void* const* d_in, const int* in_sizes, int n_in,
                              void* d_out, int out_size, void* d_ws, size_t ws_size,
                              hipStream_t stream) {
    (void)in_sizes; (void)n_in; (void)out_size; (void)ws_size;
    const float* x = (const float*)d_in[0];   // [N, D] f32
    const float* p = (const float*)d_in[1];   // [K, D] f32
    float* out = (float*)d_out;               // [N*K] T* then [1] loss

    char* ws = (char*)d_ws;
    size_t off = 0;
    auto walloc = [&](size_t bytes) -> void* {
        void* r = ws + off;
        off += (bytes + 255) & ~(size_t)255;
        return r;
    };
    _Float16* xh   = (_Float16*)walloc((size_t)NROWS * DDIM * sizeof(_Float16)); // 16 MB
    _Float16* ph   = (_Float16*)walloc((size_t)KCOLS * DDIM * sizeof(_Float16)); //  1 MB
    float*    Kmat = (float*)   walloc((size_t)NROWS * KCOLS * sizeof(float));   // 64 MB
    float*    dx   = (float*)   walloc((size_t)NROWS * sizeof(float));
    float*    dp   = (float*)   walloc((size_t)KCOLS * sizeof(float));
    float*    u    = (float*)   walloc((size_t)NROWS * sizeof(float));
    float*    v    = (float*)   walloc((size_t)KCOLS * sizeof(float));
    unsigned* cmax = (unsigned*)walloc(256);
    float*    part = (float*)   walloc((size_t)RS * KCOLS * sizeof(float));      // 256 KB
    float*    lprt = (float*)   walloc((size_t)LOSS_BLOCKS * sizeof(float));

    sk_prep_rows<<<NROWS / 8, 256, 0, stream>>>(x, xh, dx, NROWS);
    sk_prep_rows<<<KCOLS / 8, 256, 0, stream>>>(p, ph, dp, KCOLS);
    sk_init<<<NROWS / 256, 256, 0, stream>>>(u, v, cmax);

    sk_gemm_cost<<<dim3(KCOLS / 64, NROWS / 128), 256, 0, stream>>>(xh, ph, dx, dp, out, cmax);
    sk_expnorm<<<4096, 256, 0, stream>>>((float4*)out, (float4*)Kmat, cmax);

    for (int it = 0; it < 20; ++it) {
        sk_row_update<<<NROWS / 8, 256, 0, stream>>>(Kmat, v, u);
        sk_col_partial<<<dim3(KCOLS / 256, RS), 256, 0, stream>>>(Kmat, u, part);
        sk_col_finish<<<KCOLS / 256, 256, 0, stream>>>(part, v);
    }

    sk_tstar<<<LOSS_BLOCKS, 256, 0, stream>>>((float4*)out, (const float4*)Kmat, u, v, lprt);
    sk_loss_finish<<<1, 256, 0, stream>>>(lprt, out + (size_t)NROWS * KCOLS);
}
